// NerveNetMLPCriticNetwork_52003464020036
// MI455X (gfx1250) — matrix operand order repaired and checked by reading, not compiled, not run on hardware
//
#include <hip/hip_runtime.h>
#include <cstdint>
#include <cstddef>

// ---------------------------------------------------------------------------
// Types for gfx1250 WMMA
// ---------------------------------------------------------------------------
typedef __attribute__((ext_vector_type(16))) _Float16 v16h;
typedef __attribute__((ext_vector_type(8)))  float    v8f;

#define BATCH 32768
#define LOG2PI 1.8378770664093453f

// swish via hardware reciprocal (v_rcp_f32, ~1 ulp) instead of IEEE divide
__device__ __forceinline__ float swish_f(float x) {
  return x * __builtin_amdgcn_rcpf(1.0f + __expf(-x));
}

// ---------------------------------------------------------------------------
// A fragment: 16(M) x 32(K) f16 from fp32 row-major X [M x lda].
// Requires: lda % 4 == 0, rows 16B-aligned, K fully in-bounds (pre-padded).
// Contiguous-k layout used here (same as B): lane(0..15)=row M, grp=lane>>4;  K(j) = grp*16 + j
// All offsets 32-bit (buffers < 2^31 elements) -> SADDR + voffset addressing.
// ---------------------------------------------------------------------------
__device__ __forceinline__ v16h load_a_frag(const float* __restrict__ X, int lda,
                                            int m0, int kb, int lane) {
  const int mr = lane & 15;
  const int grp = lane >> 4;
  const int base = (m0 + mr) * lda + kb + grp * 16;
  const float4* p0 = (const float4*)(X + base);        // K = kb+grp*16 .. +7
  const float4* p1 = (const float4*)(X + base + 8);    // K = kb+grp*16+8 .. +7
  float4 q0 = p0[0], q1 = p0[1], q2 = p1[0], q3 = p1[1];
  v16h a;
  a[0]  = (_Float16)q0.x; a[1]  = (_Float16)q0.y; a[2]  = (_Float16)q0.z; a[3]  = (_Float16)q0.w;
  a[4]  = (_Float16)q1.x; a[5]  = (_Float16)q1.y; a[6]  = (_Float16)q1.z; a[7]  = (_Float16)q1.w;
  a[8]  = (_Float16)q2.x; a[9]  = (_Float16)q2.y; a[10] = (_Float16)q2.z; a[11] = (_Float16)q2.w;
  a[12] = (_Float16)q3.x; a[13] = (_Float16)q3.y; a[14] = (_Float16)q3.z; a[15] = (_Float16)q3.w;
  return a;
}

// ---------------------------------------------------------------------------
// B fragment: 32(K) x 16(N) f16 from pre-transposed f16 Wt [Npad x Kpad].
// Contiguous-k layout used here (same as A): lanes 0-15 hold K=0..15 (N=lane); lanes 16-31 hold K=16..31
//   -> lane reads 16 contiguous halves at Wt[n][kb + grp*16]
// ---------------------------------------------------------------------------
__device__ __forceinline__ v16h load_b_frag(const _Float16* __restrict__ Bt, int Kpad,
                                            int n0, int kb, int lane) {
  const int idx = (n0 + (lane & 15)) * Kpad + kb + (lane >> 4) * 16;
  return *(const v16h*)(Bt + idx);
}

// D layout: VGPR r -> M = m_base + r + 8*(lane>>4), N = n_base + (lane&15)
// 32-bit index stepped additively; FLAGS bit0: swish, bit1: accumulate.
template <int FLAGS>
__device__ __forceinline__ void store_frag(v8f acc, float* __restrict__ C, int ldc,
                                           int m_base, int n_base,
                                           const float* __restrict__ bias, int lane) {
  const int n = n_base + (lane & 15);
  const int mg = (lane >> 4) * 8;
  const float bv = bias[n];
  int idx = (m_base + mg) * ldc + n;
#pragma unroll
  for (int r = 0; r < 8; ++r) {
    float v = acc[r] + bv;
    if constexpr (FLAGS & 1) v = swish_f(v);
    if constexpr (FLAGS & 2) C[idx] = C[idx] + v; else C[idx] = v;
    idx += ldc;
  }
}

#define WMMA_F16(A_, B_, C_) \
  __builtin_amdgcn_wmma_f32_16x16x32_f16(false, (A_), false, (B_), (short)0, (C_), false, false)

// ---------------------------------------------------------------------------
// Wide GEMM: N multiple of 64. Block 256 thr = 8 waves (4 Mwaves x 2 Nwaves),
// block tile 128M x 64N, wave tile 32x32 (2x2 WMMA frags). No guards.
// ---------------------------------------------------------------------------
template <int FLAGS>
__global__ __launch_bounds__(256)
void wmma_gemm_wide_kernel(const float* __restrict__ A, int lda, int Kpad,
                           const _Float16* __restrict__ Bt,
                           const float* __restrict__ bias,
                           float* __restrict__ C, int ldc) {
  const int lane = threadIdx.x & 31;
  const int w    = threadIdx.x >> 5;
  const int m0   = blockIdx.x * 128 + (w >> 1) * 32;
  const int n0   = blockIdx.y * 64 + (w & 1) * 32;

  v8f acc00 = {}, acc01 = {}, acc10 = {}, acc11 = {};
  for (int kb = 0; kb < Kpad; kb += 32) {
    v16h a0 = load_a_frag(A, lda, m0,      kb, lane);
    v16h a1 = load_a_frag(A, lda, m0 + 16, kb, lane);
    v16h b0 = load_b_frag(Bt, Kpad, n0,      kb, lane);
    v16h b1 = load_b_frag(Bt, Kpad, n0 + 16, kb, lane);
    acc00 = WMMA_F16(a0, b0, acc00);
    acc10 = WMMA_F16(a1, b0, acc10);
    acc01 = WMMA_F16(a0, b1, acc01);
    acc11 = WMMA_F16(a1, b1, acc11);
  }
  store_frag<FLAGS>(acc00, C, ldc, m0,      n0,      bias, lane);
  store_frag<FLAGS>(acc10, C, ldc, m0 + 16, n0,      bias, lane);
  store_frag<FLAGS>(acc01, C, ldc, m0,      n0 + 16, bias, lane);
  store_frag<FLAGS>(acc11, C, ldc, m0 + 16, n0 + 16, bias, lane);
}

// ---------------------------------------------------------------------------
// Narrow GEMM: N == 16 (motor/head outputs padded to 16). Block 256 thr =
// 8 waves all on M, wave tile 32M x 16N, block tile 256M x 16N. No guards.
// ---------------------------------------------------------------------------
template <int FLAGS>
__global__ __launch_bounds__(256)
void wmma_gemm_n16_kernel(const float* __restrict__ A, int lda, int Kpad,
                          const _Float16* __restrict__ Bt,
                          const float* __restrict__ bias,
                          float* __restrict__ C, int ldc) {
  const int lane = threadIdx.x & 31;
  const int w    = threadIdx.x >> 5;
  const int m0   = blockIdx.x * 256 + w * 32;

  v8f acc0 = {}, acc1 = {};
  for (int kb = 0; kb < Kpad; kb += 32) {
    v16h a0 = load_a_frag(A, lda, m0,      kb, lane);
    v16h a1 = load_a_frag(A, lda, m0 + 16, kb, lane);
    v16h b0 = load_b_frag(Bt, Kpad, 0, kb, lane);
    acc0 = WMMA_F16(a0, b0, acc0);
    acc1 = WMMA_F16(a1, b0, acc1);
  }
  store_frag<FLAGS>(acc0, C, ldc, m0,      0, bias, lane);
  store_frag<FLAGS>(acc1, C, ldc, m0 + 16, 0, bias, lane);
}

// ---------------------------------------------------------------------------
// Weight pack: W [K x N] fp32 row-major -> Wt [Npad x Kpad] f16 (transposed, 0-pad)
// ---------------------------------------------------------------------------
__global__ void pack_wt_kernel(const float* __restrict__ W, _Float16* __restrict__ Wt,
                               int K, int N, int Kpad, int Npad) {
  int idx = blockIdx.x * blockDim.x + threadIdx.x;
  if (idx >= Kpad * Npad) return;
  int n = idx / Kpad;
  int k = idx - n * Kpad;
  float v = (k < K && n < N) ? W[k * N + n] : 0.0f;
  Wt[idx] = (_Float16)v;
}

__global__ void pack_heads_kernel(const float* w0, const float* w1, const float* w2,
                                  const float* b0, const float* b1, const float* b2,
                                  _Float16* __restrict__ Wt, float* __restrict__ bias) {
  int idx = blockIdx.x * blockDim.x + threadIdx.x;
  if (idx < 16 * 256) {
    int n = idx / 256;
    int k = idx - n * 256;
    float v = 0.0f;
    if (n == 0) v = w0[k]; else if (n == 1) v = w1[k]; else if (n == 2) v = w2[k];
    Wt[idx] = (_Float16)v;
  }
  if (idx < 16) {
    float v = 0.0f;
    if (idx == 0) v = b0[0]; else if (idx == 1) v = b1[0]; else if (idx == 2) v = b2[0];
    bias[idx] = v;
  }
}

__global__ void pad_bias_kernel(const float* __restrict__ src, int n, float* __restrict__ dst) {
  int i = threadIdx.x;
  if (i < 16) dst[i] = (i < n) ? src[i] : 0.0f;
}

// copy [BATCH x w] into dst rows of stride ldd at column `off`
__global__ void copy_cols_kernel(const float* __restrict__ src, int w,
                                 float* __restrict__ dst, int ldd, int off) {
  int idx = blockIdx.x * blockDim.x + threadIdx.x;
  if (idx >= BATCH * w) return;
  int r = idx / w;
  int c = idx - r * w;
  dst[r * ldd + off + c] = src[idx];
}

// ---------------------------------------------------------------------------
// Motor post-processing: actions (tanh), log-likelihood, sum(log std) partials
// ---------------------------------------------------------------------------
struct Ptr11 { const float* p[11]; };

__global__ __launch_bounds__(256)
void postprocess_kernel(const float* __restrict__ motor_buf, Ptr11 raw,
                        float* __restrict__ out, float* __restrict__ partials) {
  const int b = blockIdx.x * 256 + threadIdx.x;
  const int ACTS[11] = {6, 3, 3, 4, 4, 5, 5, 4, 4, 3, 3};
  const int AOFF[11] = {0, 6, 9, 12, 16, 20, 25, 30, 34, 38, 41};
  float loglik = 0.0f, slog = 0.0f;
#pragma unroll
  for (int k = 0; k < 11; ++k) {
    const float* mo = motor_buf + k * (BATCH * 16) + b * 16;
    const float* rw = raw.p[k] + b * ACTS[k];
    for (int j = 0; j < ACTS[k]; ++j) {
      float mean = mo[j];
      float stdp = mo[ACTS[k] + j];
      float sp   = (stdp > 0.0f) ? (stdp + log1pf(__expf(-stdp))) : log1pf(__expf(stdp));
      float std  = sp + 0.1f;
      float r    = rw[j];
      float act  = tanhf(r);
      out[b * 49 + AOFF[k] + j] = act;
      float z  = (r - mean) * __builtin_amdgcn_rcpf(std);
      float ls = __logf(std);
      loglik += -0.5f * z * z - ls - 0.5f * LOG2PI - __logf(1.0f - act * act + 1e-6f);
      slog   += ls;
    }
  }
  out[b * 49 + 44] = loglik;

  __shared__ float red[256];
  red[threadIdx.x] = slog;
  __syncthreads();
  for (int s = 128; s > 0; s >>= 1) {
    if (threadIdx.x < s) red[threadIdx.x] += red[threadIdx.x + s];
    __syncthreads();
  }
  if (threadIdx.x == 0) partials[blockIdx.x] = red[0];
}

__global__ void reduce_reg_kernel(const float* __restrict__ partials, float* __restrict__ reg) {
  float s = 0.0f;
  for (int i = 0; i < 128; ++i) s += partials[i];   // fixed order -> deterministic
  float ent = 0.5f * (1.0f + LOG2PI) * 44.0f + s / (float)BATCH;
  *reg = -0.01f * ent;
}

__global__ __launch_bounds__(256)
void finalize_kernel(const float* __restrict__ vals, const float* __restrict__ reg,
                     float* __restrict__ out) {
  int b = blockIdx.x * 256 + threadIdx.x;
  float r = *reg;
  out[b * 49 + 45] = vals[b * 16 + 0];
  out[b * 49 + 46] = vals[b * 16 + 1];
  out[b * 49 + 47] = vals[b * 16 + 2];
  out[b * 49 + 48] = r;
}

// ---------------------------------------------------------------------------
// Host launcher
// ---------------------------------------------------------------------------
extern "C" void kernel_launch(void* const* d_in, const int* in_sizes, int n_in,
                              void* d_out, int out_size, void* d_ws, size_t ws_size,
                              hipStream_t stream) {
  (void)in_sizes; (void)n_in; (void)out_size; (void)ws_size;

  static const int OBSs[11] = {60, 30, 20, 25, 25, 15, 15, 25, 25, 15, 15};
  static const int ACTs[11] = {6, 3, 3, 4, 4, 5, 5, 4, 4, 3, 3};
  static const int OOFF[11] = {0, 60, 90, 110, 135, 160, 175, 190, 215, 240, 255};

  // Input indices (setup_inputs dict insertion order, tuples flattened (W,b)):
  //  obs: 0..10, raw_action: 11..21, inp: 22+2k/23+2k, aff: 44+2k/45+2k,
  //  eff: 66+2k/67+2k, motor: 88+2k/89+2k, enc: 110..115, heads: 116..121
  const float* obs[11];
  for (int k = 0; k < 11; ++k) obs[k] = (const float*)d_in[k];
  Ptr11 raw;
  for (int k = 0; k < 11; ++k) raw.p[k] = (const float*)d_in[11 + k];

  float* out = (float*)d_out;

  // ---- workspace carve-up (256B aligned) ----
  char* wsb = (char*)d_ws;
  size_t off = 0;
  auto alloc = [&](size_t bytes) -> void* {
    off = (off + 255) & ~(size_t)255;
    void* p = wsb + off;
    off += bytes;
    return p;
  };

  int kpad_inp[11];
  size_t xpad_off[11];   // in floats within Xpad_all
  size_t xpad_total = 0;
  for (int k = 0; k < 11; ++k) {
    kpad_inp[k] = (OBSs[k] + 31) & ~31;
    xpad_off[k] = xpad_total;
    xpad_total += (size_t)BATCH * kpad_inp[k];
  }

  float* Xcat     = (float*)alloc((size_t)BATCH * 288 * 4);
  float* Xpad_all = (float*)alloc(xpad_total * 4);
  float* x[11];
  for (int k = 0; k < 11; ++k) x[k] = (float*)alloc((size_t)BATCH * 256 * 4);
  float* h1   = (float*)alloc((size_t)BATCH * 512 * 4);
  float* h2   = (float*)alloc((size_t)BATCH * 512 * 4);
  float* h3   = (float*)alloc((size_t)BATCH * 256 * 4);
  float* vals = (float*)alloc((size_t)BATCH * 16 * 4);
  float* motor_buf = (float*)alloc((size_t)11 * BATCH * 16 * 4);
  _Float16* Wt_inp[11];
  for (int k = 0; k < 11; ++k) Wt_inp[k] = (_Float16*)alloc((size_t)256 * 64 * 2);
  _Float16* Wt_aff[11];
  for (int k = 0; k < 11; ++k) Wt_aff[k] = (_Float16*)alloc((size_t)256 * 256 * 2);
  _Float16* Wt_eff[11];
  for (int k = 0; k < 11; ++k) Wt_eff[k] = (_Float16*)alloc((size_t)256 * 256 * 2);
  _Float16* Wt_motor[11];
  for (int k = 0; k < 11; ++k) Wt_motor[k] = (_Float16*)alloc((size_t)16 * 256 * 2);
  _Float16* Wt_e0 = (_Float16*)alloc((size_t)512 * 288 * 2);
  _Float16* Wt_e1 = (_Float16*)alloc((size_t)512 * 512 * 2);
  _Float16* Wt_e2 = (_Float16*)alloc((size_t)256 * 512 * 2);
  _Float16* Wt_heads = (_Float16*)alloc((size_t)16 * 256 * 2);
  float* bm_pad   = (float*)alloc((size_t)11 * 16 * 4);
  float* bh_pad   = (float*)alloc((size_t)16 * 4);
  float* partials = (float*)alloc((size_t)128 * 4);
  float* regv     = (float*)alloc(4);

  auto packwt = [&](const float* W, _Float16* Wt, int K, int N, int Kpad, int Npad) {
    int total = Kpad * Npad;
    pack_wt_kernel<<<(total + 255) / 256, 256, 0, stream>>>(W, Wt, K, N, Kpad, Npad);
  };
  auto gemm_wide = [&](const float* A, int lda, int Kpad, const _Float16* Bt, int N,
                       const float* bias, float* C, int ldc, int flags) {
    dim3 grid(BATCH / 128, N / 64);
    if (flags == 1)
      wmma_gemm_wide_kernel<1><<<grid, 256, 0, stream>>>(A, lda, Kpad, Bt, bias, C, ldc);
    else
      wmma_gemm_wide_kernel<3><<<grid, 256, 0, stream>>>(A, lda, Kpad, Bt, bias, C, ldc);
  };
  auto gemm_n16 = [&](const float* A, int lda, int Kpad, const _Float16* Bt,
                      const float* bias, float* C) {
    wmma_gemm_n16_kernel<0><<<dim3(BATCH / 256, 1), 256, 0, stream>>>(A, lda, Kpad, Bt, bias, C, 16);
  };

  // ---- prepass: pack weights/biases, zero-padded activations ----
  for (int k = 0; k < 11; ++k) {
    packwt((const float*)d_in[22 + 2 * k], Wt_inp[k], OBSs[k], 256, kpad_inp[k], 256);
    packwt((const float*)d_in[44 + 2 * k], Wt_aff[k], 256, 256, 256, 256);
    packwt((const float*)d_in[66 + 2 * k], Wt_eff[k], 256, 256, 256, 256);
    packwt((const float*)d_in[88 + 2 * k], Wt_motor[k], 256, 2 * ACTs[k], 256, 16);
    pad_bias_kernel<<<1, 32, 0, stream>>>((const float*)d_in[89 + 2 * k], 2 * ACTs[k],
                                          bm_pad + k * 16);
  }
  packwt((const float*)d_in[110], Wt_e0, 270, 512, 288, 512);
  packwt((const float*)d_in[112], Wt_e1, 512, 512, 512, 512);
  packwt((const float*)d_in[114], Wt_e2, 512, 256, 512, 256);
  pack_heads_kernel<<<16, 256, 0, stream>>>(
      (const float*)d_in[116], (const float*)d_in[118], (const float*)d_in[120],
      (const float*)d_in[117], (const float*)d_in[119], (const float*)d_in[121],
      Wt_heads, bh_pad);

  // zero-padded concat obs (critic input) and per-module padded obs (GEMM A inputs)
  hipMemsetAsync(Xcat, 0, (size_t)BATCH * 288 * 4, stream);
  hipMemsetAsync(Xpad_all, 0, xpad_total * 4, stream);
  for (int k = 0; k < 11; ++k) {
    int total = BATCH * OBSs[k];
    copy_cols_kernel<<<(total + 255) / 256, 256, 0, stream>>>(obs[k], OBSs[k], Xcat, 288, OOFF[k]);
    copy_cols_kernel<<<(total + 255) / 256, 256, 0, stream>>>(obs[k], OBSs[k],
                                                              Xpad_all + xpad_off[k],
                                                              kpad_inp[k], 0);
  }

  // ---- input layers: x[k] = swish(obs[k] @ Winp + b) ----
  for (int k = 0; k < 11; ++k)
    gemm_wide(Xpad_all + xpad_off[k], kpad_inp[k], kpad_inp[k], Wt_inp[k], 256,
              (const float*)d_in[23 + 2 * k], x[k], 256, /*swish*/1);

  // module indices: 0 root,1 torso,2 head,3 arm_L,4 arm_R,5 hand_L,6 hand_R,
  //                 7 leg_L,8 leg_R,9 foot_L,10 foot_R
  auto AFF = [&](int src, int dst) {
    gemm_wide(x[src], 256, 256, Wt_aff[src], 256, (const float*)d_in[45 + 2 * src],
              x[dst], 256, /*swish|acc*/3);
  };
  auto EFF = [&](int key, int src, int dst) {
    gemm_wide(x[src], 256, 256, Wt_eff[key], 256, (const float*)d_in[67 + 2 * key],
              x[dst], 256, /*swish|acc*/3);
  };

  // afferent (leaf -> root), exact reference order
  AFF(5, 3);  AFF(6, 4);  AFF(9, 7);  AFF(10, 8);
  AFF(3, 0);  AFF(4, 0);  AFF(7, 0);  AFF(8, 0);  AFF(1, 0);  AFF(2, 0);

  // efferent (root -> leaf), exact reference order
  EFF(2, 0, 2);   EFF(1, 0, 1);
  EFF(3, 0, 3);   EFF(5, 3, 5);
  EFF(4, 0, 4);   EFF(6, 4, 6);
  EFF(7, 0, 7);   EFF(9, 7, 9);
  EFF(8, 0, 8);   EFF(10, 8, 10);

  // motor heads (linear, N padded to 16)
  for (int k = 0; k < 11; ++k)
    gemm_n16(x[k], 256, 256, Wt_motor[k], bm_pad + k * 16,
             motor_buf + (size_t)k * BATCH * 16);

  // tanh-Normal sampler epilogue: actions, loglik, sum(log std) partials
  postprocess_kernel<<<BATCH / 256, 256, 0, stream>>>(motor_buf, raw, out, partials);
  reduce_reg_kernel<<<1, 1, 0, stream>>>(partials, regv);

  // critic: encoder + heads (Xcat zero-padded -> Kact==Kpad==288 is exact)
  gemm_wide(Xcat, 288, 288, Wt_e0, 512, (const float*)d_in[111], h1, 512, 1);
  gemm_wide(h1, 512, 512, Wt_e1, 512, (const float*)d_in[113], h2, 512, 1);
  gemm_wide(h2, 512, 512, Wt_e2, 256, (const float*)d_in[115], h3, 256, 1);
  gemm_n16(h3, 256, 256, Wt_heads, bh_pad, vals);

  // values + reg columns
  finalize_kernel<<<BATCH / 256, 256, 0, stream>>>(vals, regv, out);
}
